// FB_LiDiff_Transformer_17952963298121
// MI455X (gfx1250) — compile-verified, matrix operand order checked by reading, not run on hardware
//
#include <hip/hip_runtime.h>
#include <hip/hip_bf16.h>

// ---------------------------------------------------------------------------
// Spiking transformer block forward, MI455X (gfx1250), wave32 + WMMA bf16.
// T=4 B=2 C=384 N=4096 heads=8 d=48 hidden=1536
// GEMM A-tile staging uses the Tensor Data Mover (tensor_load_to_lds) when the
// builtin is available; manual staging otherwise.
// ---------------------------------------------------------------------------

typedef __bf16 bf16;
typedef __attribute__((ext_vector_type(16))) __bf16 bf16x16;
typedef __attribute__((ext_vector_type(8)))  float  floatx8;
typedef __attribute__((ext_vector_type(4)))  unsigned int uint32x4;
typedef __attribute__((ext_vector_type(8)))  int int32x8;
typedef __attribute__((ext_vector_type(4)))  int int32x4;

#define GN   4096          // N = H*W
#define CDIM 384
#define CHID 1536
#define NSLICE 8           // T*B
#define NHEAD 8
#define HD   48

#if defined(__gfx1250__) && __has_builtin(__builtin_amdgcn_tensor_load_to_lds) && \
    __has_builtin(__builtin_amdgcn_s_wait_tensorcnt)
#define USE_TDM 1
#else
#define USE_TDM 0
#endif

static __device__ __forceinline__ floatx8 wmma_bf16(bf16x16 a, bf16x16 b, floatx8 c) {
  return __builtin_amdgcn_wmma_f32_16x16x32_bf16(false, a, false, b, (short)0, c, false, false);
}

// Fragment load for the CDNA5 16-bit A/B layout (16x32 / 32x16):
//   e0..7 -> k = kSel + e, e8..15 -> k = kSel + 16 + (e-8), kSel = (L&16)?8:0.
// Two contiguous 16-byte loads from a k-major row.
static __device__ __forceinline__ bf16x16 load_frag(const bf16* rowBase, int kSel) {
  union { uint4 u[2]; bf16x16 v; } c;
  c.u[0] = *reinterpret_cast<const uint4*>(rowBase + kSel);
  c.u[1] = *reinterpret_cast<const uint4*>(rowBase + kSel + 16);
  return c.v;
}

static __device__ __forceinline__ floatx8 zero8() {
  floatx8 z;
#pragma unroll
  for (int r = 0; r < 8; ++r) z[r] = 0.0f;
  return z;
}

#if USE_TDM
// ---------------------------------------------------------------------------
// TDM descriptor (ISA cdna5 ch.8): 2D tile load, 128 rows x 32 bf16,
// global row stride = K elements, LDS padded +8 bf16 per 32-bf16 row
// (pad_interval code 3 = 16 DWORDs, pad_amount code 3 = 4 DWORDs) => LDS
// row stride 40 bf16, matching the fragment-load layout below.
// Issued by one wave; TENSORcnt waited by that wave before the block barrier.
// ---------------------------------------------------------------------------
static __device__ __forceinline__ void tdm_load_a_tile(const bf16* gsrc,
                                                       unsigned lds_byte, int K) {
  const unsigned long long ga = (unsigned long long)(uintptr_t)gsrc;
  uint32x4 g0;
  g0[0] = 1u;                                   // count=1 user descriptor
  g0[1] = lds_byte;                             // lds_addr
  g0[2] = (unsigned)(ga & 0xffffffffu);         // global_addr[31:0]
  g0[3] = (unsigned)((ga >> 32) & 0x01ffffffu)  // global_addr[56:32]
          | (2u << 30);                         // type = 2 (image)
  const unsigned TD0 = 0x7fffffffu;             // huge tensor dims: no OOB clamp
  const unsigned TD1 = 0x000fffffu;
  int32x8 g1;
  g1[0] = (int)((1u << 16)        // data_size = 2 bytes
                | (1u << 20)      // pad_enable
                | (3u << 22)      // pad_interval: 16 DWORDs (one 32-bf16 row)
                | (3u << 25));    // pad_amount: 4 DWORDs (8 bf16)
  g1[1] = (int)((TD0 & 0xffffu) << 16);                          // tensor_dim0 lo
  g1[2] = (int)(((TD0 >> 16) & 0xffffu) | ((TD1 & 0xffffu) << 16)); // dim0 hi | dim1 lo
  g1[3] = (int)(((TD1 >> 16) & 0xffffu) | (32u << 16));          // dim1 hi | tile_dim0=32
  g1[4] = (int)128u;                                             // tile_dim1=128, tile_dim2=0
  g1[5] = (int)(unsigned)K;                                      // tensor_dim0_stride lo32
  g1[6] = 0;                                                     // stride hi | dim1_stride lo
  g1[7] = 0;
  int32x4 gz; gz[0] = 0; gz[1] = 0; gz[2] = 0; gz[3] = 0;        // groups 2/3: 2D tensor
#if __clang_major__ >= 23
  int32x8 z8;
#pragma unroll
  for (int i = 0; i < 8; ++i) z8[i] = 0;
  __builtin_amdgcn_tensor_load_to_lds(g0, g1, gz, gz, z8, 0);
#else
  __builtin_amdgcn_tensor_load_to_lds(g0, g1, gz, gz, 0);
#endif
}
#endif  // USE_TDM

// ---------------------------------------------------------------------------
// fp32 -> bf16 convert
// ---------------------------------------------------------------------------
__global__ __launch_bounds__(256) void cvt_f32_bf16(const float* __restrict__ src,
                                                    bf16* __restrict__ dst, size_t n) {
  size_t i = (size_t)blockIdx.x * 256 + threadIdx.x;
  if (i < n) dst[i] = (bf16)src[i];
}

// ---------------------------------------------------------------------------
// BN coefficients: scale = g/sqrt(v+eps), shift = b + (bias - m)*scale
// ---------------------------------------------------------------------------
__global__ __launch_bounds__(256) void bncoef(const float* __restrict__ g,
                                              const float* __restrict__ bb,
                                              const float* __restrict__ m,
                                              const float* __restrict__ vv,
                                              const float* __restrict__ bias,
                                              float* __restrict__ sc,
                                              float* __restrict__ sh, int n) {
  int i = blockIdx.x * 256 + threadIdx.x;
  if (i < n) {
    float s = g[i] * rsqrtf(vv[i] + 1e-5f);
    sc[i] = s;
    float bi = bias ? bias[i] : 0.0f;
    sh[i] = bb[i] + (bi - m[i]) * s;
  }
}

// ---------------------------------------------------------------------------
// Tiled bf16 GEMM: Y[s][M,GN] = W[M,K] * ACT[s][K,GN]   (s = 0..7 slices)
// Block = 256 threads = 8 waves, 128x128 output tile, K-step 32.
// Wave (wm,wn) in 4x2 grid computes 32x64 -> 2x4 WMMA tiles.
// A tile staged by TDM (tensor_load_to_lds w/ LDS pad) when available.
// ---------------------------------------------------------------------------
__global__ __launch_bounds__(256) void gemm_bf16(const bf16* __restrict__ W,
                                                 const bf16* __restrict__ act,
                                                 float* __restrict__ Y,
                                                 int M, int K) {
  const int slice = blockIdx.z;
  const int mTile = blockIdx.y;
  const int nTile = blockIdx.x;

  const bf16* A = W + (size_t)mTile * 128 * K;
  const bf16* X = act + (size_t)slice * K * GN + (size_t)nTile * 128;
  float* Yo = Y + (size_t)slice * M * GN + (size_t)mTile * 128 * GN + (size_t)nTile * 128;

  __shared__ __align__(16) bf16 lA[128 * 40];   // [m][k], stride 40 (80B, 16B aligned)
  __shared__ __align__(16) bf16 lB[128 * 40];   // [n][k] transposed, stride 40

  const int tid  = threadIdx.x;
  const int wave = tid >> 5;
  const int lane = tid & 31;
  const int wm   = wave & 3;   // m offset = 32*wm
  const int wn   = wave >> 2;  // n offset = 64*wn
  const int rowL = lane & 15;
  const int kSel = (lane & 16) ? 8 : 0;

  floatx8 acc[2][4];
#pragma unroll
  for (int i = 0; i < 2; ++i)
#pragma unroll
    for (int j = 0; j < 4; ++j) acc[i][j] = zero8();

  for (int k0 = 0; k0 < K; k0 += 32) {
#if USE_TDM
    // --- stage A tile via Tensor Data Mover (one descriptor per workgroup) ---
    if (tid == 0)
      tdm_load_a_tile(A + k0, (unsigned)(uintptr_t)(void*)&lA[0], K);
#else
    // --- stage A tile (128 x 32): thread -> row tid>>1, half tid&1 ---
    {
      const int r = tid >> 1, half = tid & 1;
      const uint4* g = reinterpret_cast<const uint4*>(A + (size_t)r * K + k0 + half * 16);
      uint4 d0 = g[0], d1 = g[1];
      if (k0 + 32 < K) __builtin_prefetch(A + (size_t)r * K + k0 + 32, 0, 1);
      uint4* s = reinterpret_cast<uint4*>(&lA[r * 40 + half * 16]);
      s[0] = d0; s[1] = d1;
    }
#endif
    // --- stage B tile (32 x 128) transposed into [n][k] ---
    {
      const int kk = tid & 31;
      const int nC = tid >> 5;  // 0..7, 16-wide n chunk
      const uint4* gv = reinterpret_cast<const uint4*>(X + (size_t)(k0 + kk) * GN + nC * 16);
      union { uint4 u[2]; bf16 h[16]; } uu;
      uu.u[0] = gv[0]; uu.u[1] = gv[1];
      if (k0 + 32 < K) __builtin_prefetch(X + (size_t)(k0 + 32 + kk) * GN + nC * 16, 0, 1);
#pragma unroll
      for (int j = 0; j < 16; ++j) lB[(nC * 16 + j) * 40 + kk] = uu.h[j];
    }
#if USE_TDM
    if (wave == 0) __builtin_amdgcn_s_wait_tensorcnt(0);
#endif
    __syncthreads();

    bf16x16 af[2], bfr[4];
#pragma unroll
    for (int i = 0; i < 2; ++i)
      af[i] = load_frag(&lA[(wm * 32 + i * 16 + rowL) * 40], kSel);
#pragma unroll
    for (int j = 0; j < 4; ++j)
      bfr[j] = load_frag(&lB[(wn * 64 + j * 16 + rowL) * 40], kSel);

#pragma unroll
    for (int i = 0; i < 2; ++i)
#pragma unroll
      for (int j = 0; j < 4; ++j)
        acc[i][j] = wmma_bf16(af[i], bfr[j], acc[i][j]);
    __syncthreads();
  }

  // D layout: vgpr r -> M = r + ((lane&16)?8:0), N = lane&15
  const int mHi = (lane >> 4) << 3;
#pragma unroll
  for (int i = 0; i < 2; ++i)
#pragma unroll
    for (int j = 0; j < 4; ++j)
#pragma unroll
      for (int r = 0; r < 8; ++r)
        Yo[(size_t)(wm * 32 + i * 16 + mHi + r) * GN + wn * 64 + j * 16 + rowL] = acc[i][j][r];
}

// ---------------------------------------------------------------------------
// kv[s,h][d,e] = sum_n ks[s, h*48+d, n] * vs[s, h*48+e, n]
// one wave per 16x16 tile, 3x3 tiles per head; fragments direct from global.
// ---------------------------------------------------------------------------
__global__ __launch_bounds__(32) void kv_kernel(const bf16* __restrict__ ks,
                                                const bf16* __restrict__ vs,
                                                float* __restrict__ kv) {
  int bid = blockIdx.x;
  const int ni = bid % 3; bid /= 3;
  const int mi = bid % 3; bid /= 3;
  const int h  = bid % NHEAD; bid /= NHEAD;
  const int slice = bid;  // 0..7

  const bf16* Kp = ks + (size_t)slice * CDIM * GN + (size_t)(h * HD + mi * 16) * GN;
  const bf16* Vp = vs + (size_t)slice * CDIM * GN + (size_t)(h * HD + ni * 16) * GN;

  const int lane = threadIdx.x & 31;
  const int rowL = lane & 15;
  const int kSel = (lane & 16) ? 8 : 0;

  floatx8 acc = zero8();
  for (int n0 = 0; n0 < GN; n0 += 32) {
    bf16x16 a = load_frag(Kp + (size_t)rowL * GN + n0, kSel);
    bf16x16 b = load_frag(Vp + (size_t)rowL * GN + n0, kSel);
    acc = wmma_bf16(a, b, acc);
  }

  float* out = kv + ((size_t)slice * NHEAD + h) * (HD * HD);
  const int mHi = (lane >> 4) << 3;
#pragma unroll
  for (int r = 0; r < 8; ++r)
    out[(size_t)(mi * 16 + mHi + r) * HD + ni * 16 + rowL] = acc[r];
}

// ---------------------------------------------------------------------------
// o_pre[s, h*48+e, n] = sum_f kv[s,h][f,e] * qs[s, h*48+f, n]
// A = kv^T (48x48 zero-padded to K=64) in LDS; B = q tile transposed in LDS.
// ---------------------------------------------------------------------------
__global__ __launch_bounds__(256) void attn_o_kernel(const float* __restrict__ kv,
                                                     const bf16* __restrict__ qs,
                                                     float* __restrict__ Y) {
  const int nTile = blockIdx.x;   // 0..31 (128 cols)
  const int h     = blockIdx.y;   // 0..7
  const int slice = blockIdx.z;   // 0..7

  __shared__ __align__(16) bf16 lA[HD * 72];    // A[e][f], f padded to 64, stride 72
  __shared__ __align__(16) bf16 lB[128 * 72];   // B[n][f], f padded to 64, stride 72

  const float* kvp = kv + ((size_t)slice * NHEAD + h) * (HD * HD);
  const bf16*  Q   = qs + (size_t)slice * CDIM * GN + (size_t)h * HD * GN + (size_t)nTile * 128;
  float*       Yo  = Y  + (size_t)slice * CDIM * GN + (size_t)h * HD * GN + (size_t)nTile * 128;

  const int tid = threadIdx.x;

  for (int i = tid; i < HD * 16; i += 256) {       // zero f = 48..63 pad (A)
    int e = i >> 4, f = 48 + (i & 15);
    lA[e * 72 + f] = (bf16)0.0f;
  }
  for (int i = tid; i < 128 * 16; i += 256) {      // zero f = 48..63 pad (B)
    int n = i >> 4, f = 48 + (i & 15);
    lB[n * 72 + f] = (bf16)0.0f;
  }
  for (int i = tid; i < HD * HD; i += 256) {       // A = kv^T
    int f = i / HD, e = i % HD;
    lA[e * 72 + f] = (bf16)kvp[f * HD + e];
  }
  for (int i = tid; i < HD * 8; i += 256) {        // B: q rows transposed
    int f = i % HD, nC = i / HD;
    const uint4* gv = reinterpret_cast<const uint4*>(Q + (size_t)f * GN + nC * 16);
    union { uint4 u[2]; bf16 hh[16]; } uu;
    uu.u[0] = gv[0]; uu.u[1] = gv[1];
#pragma unroll
    for (int j = 0; j < 16; ++j) lB[(nC * 16 + j) * 72 + f] = uu.hh[j];
  }
  __syncthreads();

  const int wave = tid >> 5;
  const int lane = tid & 31;
  const int rowL = lane & 15;
  const int kSel = (lane & 16) ? 8 : 0;

  floatx8 acc[3];
#pragma unroll
  for (int mi = 0; mi < 3; ++mi) acc[mi] = zero8();

#pragma unroll
  for (int k0 = 0; k0 < 64; k0 += 32) {
    bf16x16 bfr = load_frag(&lB[(wave * 16 + rowL) * 72 + k0], kSel);
#pragma unroll
    for (int mi = 0; mi < 3; ++mi) {
      bf16x16 afr = load_frag(&lA[(mi * 16 + rowL) * 72 + k0], kSel);
      acc[mi] = wmma_bf16(afr, bfr, acc[mi]);
    }
  }

  const int mHi = (lane >> 4) << 3;
#pragma unroll
  for (int mi = 0; mi < 3; ++mi)
#pragma unroll
    for (int r = 0; r < 8; ++r)
      Yo[(size_t)(mi * 16 + mHi + r) * GN + wave * 16 + rowL] = acc[mi][r];
}

// ---------------------------------------------------------------------------
// Fused BN + LIF scan over T=4 (tau=2, hard reset, detach).
//   z_t = y_t*scale + shift ; v = (v+z)/2 ; s = (v>=vth) ; v *= (1-s)
// Nullable outputs: sOut (bf16 spikes), addOut = addIn + s (fp32).
// ---------------------------------------------------------------------------
__global__ __launch_bounds__(256) void lif_kernel(const float* __restrict__ y,
                                                  const float* __restrict__ sc,
                                                  const float* __restrict__ sh,
                                                  float cscale, float vth,
                                                  bf16* __restrict__ sOut,
                                                  const float* __restrict__ addIn,
                                                  float* __restrict__ addOut,
                                                  int Cc) {
  const size_t P = (size_t)2 * Cc * GN;  // per-timestep elements (B*Cc*N)
  size_t idx = (size_t)blockIdx.x * 256 + threadIdx.x;
  if (idx >= P) return;
  const int c = (int)((idx / GN) % Cc);
  const float a = sc ? sc[c] : cscale;
  const float b = sh ? sh[c] : 0.0f;

  float v = 0.0f;
#pragma unroll
  for (int t = 0; t < 4; ++t) {
    const size_t o = (size_t)t * P + idx;
    float z = y[o] * a + b;
    v = 0.5f * (v + z);
    float s = (v >= vth) ? 1.0f : 0.0f;
    if (sOut)   sOut[o]   = (bf16)s;
    if (addOut) addOut[o] = addIn[o] + s;
    v *= (1.0f - s);
  }
}

// ---------------------------------------------------------------------------
// Host launcher
// ---------------------------------------------------------------------------
extern "C" void kernel_launch(void* const* d_in, const int* in_sizes, int n_in,
                              void* d_out, int out_size, void* d_ws, size_t ws_size,
                              hipStream_t stream) {
  const float* x      = (const float*)d_in[0];
  const float* qw     = (const float*)d_in[1];
  const float* kw     = (const float*)d_in[6];
  const float* vw     = (const float*)d_in[11];
  const float* projw  = (const float*)d_in[16];
  const float* projb  = (const float*)d_in[17];
  const float* mlp1w  = (const float*)d_in[22];
  const float* mlp1b  = (const float*)d_in[23];
  const float* mlp2w  = (const float*)d_in[28];
  const float* mlp2b  = (const float*)d_in[29];

  const size_t ACT  = (size_t)NSLICE * CDIM * GN;   // 12,582,912
  const size_t ACTH = (size_t)NSLICE * CHID * GN;   // 50,331,648

  char* w = (char*)d_ws;
  auto alloc = [&](size_t bytes) -> char* {
    char* p = w;
    w += (bytes + 255) & ~(size_t)255;
    return p;
  };

  bf16* xb   = (bf16*)alloc(ACT * 2);            // x (then reused as x1 bf16)
  bf16* wq_b = (bf16*)alloc((size_t)CDIM * CDIM * 2);
  bf16* wk_b = (bf16*)alloc((size_t)CDIM * CDIM * 2);
  bf16* wv_b = (bf16*)alloc((size_t)CDIM * CDIM * 2);
  bf16* wp_b = (bf16*)alloc((size_t)CDIM * CDIM * 2);
  bf16* w1_b = (bf16*)alloc((size_t)CHID * CDIM * 2);
  bf16* w2_b = (bf16*)alloc((size_t)CDIM * CHID * 2);
  float* sc_q = (float*)alloc(CDIM * 4); float* sh_q = (float*)alloc(CDIM * 4);
  float* sc_k = (float*)alloc(CDIM * 4); float* sh_k = (float*)alloc(CDIM * 4);
  float* sc_v = (float*)alloc(CDIM * 4); float* sh_v = (float*)alloc(CDIM * 4);
  float* sc_p = (float*)alloc(CDIM * 4); float* sh_p = (float*)alloc(CDIM * 4);
  float* sc_1 = (float*)alloc(CHID * 4); float* sh_1 = (float*)alloc(CHID * 4);
  float* sc_2 = (float*)alloc(CDIM * 4); float* sh_2 = (float*)alloc(CDIM * 4);
  bf16* qs  = (bf16*)alloc(ACT * 2);
  bf16* ksb = (bf16*)alloc(ACT * 2);             // k spikes, reused as attn-out spikes
  bf16* vsb = (bf16*)alloc(ACT * 2);
  float* kvb = (float*)alloc((size_t)NSLICE * NHEAD * HD * HD * 4);
  float* x1  = (float*)alloc(ACT * 4);
  bf16* hsb  = (bf16*)alloc(ACTH * 2);
  float* ybuf = (float*)alloc(ACTH * 4);
  (void)ws_size; (void)in_sizes; (void)n_in; (void)out_size;

  const dim3 B256(256);
  auto gblk = [](size_t n) { return dim3((unsigned)((n + 255) / 256)); };

  // --- precision staging ---
  cvt_f32_bf16<<<gblk(ACT), B256, 0, stream>>>(x, xb, ACT);
  cvt_f32_bf16<<<gblk((size_t)CDIM * CDIM), B256, 0, stream>>>(qw, wq_b, (size_t)CDIM * CDIM);
  cvt_f32_bf16<<<gblk((size_t)CDIM * CDIM), B256, 0, stream>>>(kw, wk_b, (size_t)CDIM * CDIM);
  cvt_f32_bf16<<<gblk((size_t)CDIM * CDIM), B256, 0, stream>>>(vw, wv_b, (size_t)CDIM * CDIM);
  cvt_f32_bf16<<<gblk((size_t)CDIM * CDIM), B256, 0, stream>>>(projw, wp_b, (size_t)CDIM * CDIM);
  cvt_f32_bf16<<<gblk((size_t)CHID * CDIM), B256, 0, stream>>>(mlp1w, w1_b, (size_t)CHID * CDIM);
  cvt_f32_bf16<<<gblk((size_t)CDIM * CHID), B256, 0, stream>>>(mlp2w, w2_b, (size_t)CDIM * CHID);

  // --- BN coefficients (conv bias folded into shift) ---
  bncoef<<<gblk(CDIM), B256, 0, stream>>>((const float*)d_in[2], (const float*)d_in[3],
                                          (const float*)d_in[4], (const float*)d_in[5],
                                          nullptr, sc_q, sh_q, CDIM);
  bncoef<<<gblk(CDIM), B256, 0, stream>>>((const float*)d_in[7], (const float*)d_in[8],
                                          (const float*)d_in[9], (const float*)d_in[10],
                                          nullptr, sc_k, sh_k, CDIM);
  bncoef<<<gblk(CDIM), B256, 0, stream>>>((const float*)d_in[12], (const float*)d_in[13],
                                          (const float*)d_in[14], (const float*)d_in[15],
                                          nullptr, sc_v, sh_v, CDIM);
  bncoef<<<gblk(CDIM), B256, 0, stream>>>((const float*)d_in[18], (const float*)d_in[19],
                                          (const float*)d_in[20], (const float*)d_in[21],
                                          projb, sc_p, sh_p, CDIM);
  bncoef<<<gblk(CHID), B256, 0, stream>>>((const float*)d_in[24], (const float*)d_in[25],
                                          (const float*)d_in[26], (const float*)d_in[27],
                                          mlp1b, sc_1, sh_1, CHID);
  bncoef<<<gblk(CDIM), B256, 0, stream>>>((const float*)d_in[30], (const float*)d_in[31],
                                          (const float*)d_in[32], (const float*)d_in[33],
                                          mlp2b, sc_2, sh_2, CDIM);

  const dim3 gemmGridC(32, CDIM / 128, NSLICE);   // M=384
  const dim3 gemmGridH(32, CHID / 128, NSLICE);   // M=1536
  const size_t PC = (size_t)2 * CDIM * GN;
  const size_t PH = (size_t)2 * CHID * GN;

  // --- q / k / v branches: GEMM -> BN+LIF -> spikes ---
  gemm_bf16<<<gemmGridC, B256, 0, stream>>>(wq_b, xb, ybuf, CDIM, CDIM);
  lif_kernel<<<gblk(PC), B256, 0, stream>>>(ybuf, sc_q, sh_q, 1.0f, 1.0f, qs, nullptr, nullptr, CDIM);
  gemm_bf16<<<gemmGridC, B256, 0, stream>>>(wk_b, xb, ybuf, CDIM, CDIM);
  lif_kernel<<<gblk(PC), B256, 0, stream>>>(ybuf, sc_k, sh_k, 1.0f, 1.0f, ksb, nullptr, nullptr, CDIM);
  gemm_bf16<<<gemmGridC, B256, 0, stream>>>(wv_b, xb, ybuf, CDIM, CDIM);
  lif_kernel<<<gblk(PC), B256, 0, stream>>>(ybuf, sc_v, sh_v, 1.0f, 1.0f, vsb, nullptr, nullptr, CDIM);

  // --- linear attention ---
  kv_kernel<<<dim3(NSLICE * NHEAD * 9), dim3(32), 0, stream>>>(ksb, vsb, kvb);
  attn_o_kernel<<<dim3(32, NHEAD, NSLICE), B256, 0, stream>>>(kvb, qs, ybuf);
  // attn LIF: z = 0.125*o, vth = 0.5, spikes -> reuse ksb
  lif_kernel<<<gblk(PC), B256, 0, stream>>>(ybuf, nullptr, nullptr, 0.125f, 0.5f, ksb, nullptr, nullptr, CDIM);

  // --- proj: GEMM -> BN+LIF -> x1 = x + s (fp32), then x1 -> bf16 (reuse xb) ---
  gemm_bf16<<<gemmGridC, B256, 0, stream>>>(wp_b, ksb, ybuf, CDIM, CDIM);
  lif_kernel<<<gblk(PC), B256, 0, stream>>>(ybuf, sc_p, sh_p, 1.0f, 1.0f, nullptr, x, x1, CDIM);
  cvt_f32_bf16<<<gblk(ACT), B256, 0, stream>>>(x1, xb, ACT);

  // --- MLP ---
  gemm_bf16<<<gemmGridH, B256, 0, stream>>>(w1_b, xb, ybuf, CHID, CDIM);
  lif_kernel<<<gblk(PH), B256, 0, stream>>>(ybuf, sc_1, sh_1, 1.0f, 1.0f, hsb, nullptr, nullptr, CHID);
  gemm_bf16<<<gemmGridC, B256, 0, stream>>>(w2_b, hsb, ybuf, CDIM, CHID);
  lif_kernel<<<gblk(PC), B256, 0, stream>>>(ybuf, sc_2, sh_2, 1.0f, 1.0f, nullptr, x1, (float*)d_out, CDIM);
}
